// PathwayAttentionEncoder_54735063220352
// MI455X (gfx1250) — compile-verified, matrix operand order checked by reading, not run on hardware
//
#include <hip/hip_runtime.h>
#include <hip/hip_bf16.h>

// Problem constants (from reference)
#define BB 2048
#define NG 20000
#define PP 256
#define GG 64
#define EE 128
#define TT 32
#define HH 4
#define HD 32

typedef __attribute__((ext_vector_type(16))) _Float16 v16h;
typedef __attribute__((ext_vector_type(8)))  _Float16 h8;
typedef __attribute__((ext_vector_type(8)))  float    v8f;

__device__ __forceinline__ v8f wmma16x16x32(v16h a, v16h b, v8f c) {
  return __builtin_amdgcn_wmma_f32_16x16x32_f16(
      /*neg_a=*/false, a, /*neg_b=*/false, b,
      /*c_mod=*/(short)0, c, /*reuse_a=*/false, /*reuse_b=*/false);
}

union Frag16 { v16h v; h8 h[2]; };

// ---------------------------------------------------------------------------
// K1: tokens[b,p,:] = relu(xg[b,p,:] @ Wp[p] + bp[p]),   xg = gather(x_genes)
// grid (B/16, P), block 128 (4 waves). Each wave owns 32 output columns.
// A tile row-major (M x K), B tile stored TRANSPOSED (N x K) in LDS so that
// each lane's WMMA fragment = two contiguous 16B chunks -> ds_load_b128.
// Fragment layout (CDNA5 ISA §7.12.2): lane half picks +8 K offset,
// v16h elems [0..7] = K(kb..kb+7), [8..15] = K(kb+16..kb+23).
// ---------------------------------------------------------------------------
__global__ void k_tokens(const float* __restrict__ x_genes,
                         const int*   __restrict__ pathway_idx,
                         const float* __restrict__ Wp,
                         const float* __restrict__ bp,
                         _Float16*    __restrict__ tok) {
  __shared__ int sIdx[GG];
  __shared__ __align__(16) _Float16 sXg[16][GG + 8];   // 16 x 72 halves
  __shared__ __align__(16) _Float16 sWpT[EE][GG + 8];  // 128 x 72 halves (N-major)

  const int tid = threadIdx.x;
  const int p  = blockIdx.y;
  const int m0 = blockIdx.x * 16;

  if (tid < GG) sIdx[tid] = pathway_idx[p * GG + tid];
  __syncthreads();

  // gather A tile: 16 rows x 64 gathered genes, f32 -> f16
  for (int i = tid; i < 16 * GG; i += 128) {
    int m = i >> 6, g = i & (GG - 1);
    sXg[m][g] = (_Float16)x_genes[(size_t)(m0 + m) * NG + sIdx[g]];
  }
  // B tile: Wp[p] is (g,e) row-major in global; store transposed (e,g) in LDS.
  const float* wp = Wp + (size_t)p * GG * EE;
  for (int i = tid; i < GG * EE; i += 128) {
    int g = i >> 7, e = i & (EE - 1);      // coalesced global read
    sWpT[e][g] = (_Float16)wp[i];
  }
  __syncthreads();

  const int wv   = tid >> 5;         // wave id 0..3 -> columns [wv*32, wv*32+32)
  const int lane = tid & 31;
  const int half = lane >> 4;        // lane group: +8 K offset per ISA layout
  const int lo   = lane & 15;        // M for A, N for B/D
  const int nb0  = wv * 32, nb1 = wv * 32 + 16;

  v8f acc0 = {}; v8f acc1 = {};
#pragma unroll
  for (int s = 0; s < 2; ++s) {      // two K=32 steps
    const int kb = 8 * half + 32 * s;
    Frag16 a, b0, b1;
    a.h[0]  = *(const h8*)&sXg[lo][kb];
    a.h[1]  = *(const h8*)&sXg[lo][kb + 16];
    b0.h[0] = *(const h8*)&sWpT[nb0 + lo][kb];
    b0.h[1] = *(const h8*)&sWpT[nb0 + lo][kb + 16];
    b1.h[0] = *(const h8*)&sWpT[nb1 + lo][kb];
    b1.h[1] = *(const h8*)&sWpT[nb1 + lo][kb + 16];
    acc0 = wmma16x16x32(a.v, b0.v, acc0);
    acc1 = wmma16x16x32(a.v, b1.v, acc1);
  }

  // D layout: N = lo, M = r + 8*half. Apply bias + relu, store f16 tokens.
  const float* bpp = bp + p * EE;
  float bia0 = bpp[nb0 + lo], bia1 = bpp[nb1 + lo];
#pragma unroll
  for (int r = 0; r < 8; ++r) {
    int m = r + 8 * half;
    size_t base = ((size_t)(m0 + m) * PP + p) * EE;
    tok[base + nb0 + lo] = (_Float16)fmaxf(acc0[r] + bia0, 0.f);
    tok[base + nb1 + lo] = (_Float16)fmaxf(acc1[r] + bia1, 0.f);
  }
}

// ---------------------------------------------------------------------------
// K2: tmean[b,e] = mean_p tokens[b,p,e]   (coalesced across threads)
// ---------------------------------------------------------------------------
__global__ void k_tmean(const _Float16* __restrict__ tok,
                        float* __restrict__ tmean) {
  int b = blockIdx.x, e = threadIdx.x;
  const _Float16* t = tok + (size_t)b * PP * EE + e;
  float s = 0.f;
  for (int p = 0; p < PP; ++p) s += (float)t[(size_t)p * EE];
  tmean[b * EE + e] = s * (1.f / PP);
}

// ---------------------------------------------------------------------------
// K3: query = relu(concat(temb, tmean) @ Wq + bq)
// ---------------------------------------------------------------------------
__global__ void k_query(const float* __restrict__ tmean,
                        const float* __restrict__ tissue_emb,
                        const int*   __restrict__ x_tissue,
                        const float* __restrict__ Wq,
                        const float* __restrict__ bq,
                        float* __restrict__ query) {
  __shared__ float comb[2 * EE];
  int b = blockIdx.x, e = threadIdx.x;
  comb[e]      = tissue_emb[x_tissue[b] * EE + e];
  comb[EE + e] = tmean[b * EE + e];
  __syncthreads();
  float s = bq[e];
  for (int j = 0; j < 2 * EE; ++j) s += comb[j] * Wq[j * EE + e];
  query[b * EE + e] = fmaxf(s, 0.f);
}

// ---------------------------------------------------------------------------
// K4: scores = q.tokens/sqrt(E); softmax over P; context = aw @ tokens
// grid B, block 256 (one thread per pathway for scores; b128 token reads).
// ---------------------------------------------------------------------------
__global__ void k_attnP(const _Float16* __restrict__ tok,
                        const float* __restrict__ query,
                        float* __restrict__ context) {
  __shared__ float sq[EE];
  __shared__ float sc[PP];
  __shared__ float red[PP];
  int b = blockIdx.x, tid = threadIdx.x;
  if (tid < EE) sq[tid] = query[b * EE + tid];
  __syncthreads();

  {
    const h8* t = (const h8*)(tok + ((size_t)b * PP + tid) * EE);
    float s = 0.f;
    for (int i = 0; i < EE / 8; ++i) {
      h8 x = t[i];
#pragma unroll
      for (int j = 0; j < 8; ++j) s += sq[i * 8 + j] * (float)x[j];
    }
    sc[tid] = s * 0.08838834764831845f;  // 1/sqrt(128)
  }
  __syncthreads();

  red[tid] = sc[tid]; __syncthreads();
  for (int st = PP / 2; st > 0; st >>= 1) {
    if (tid < st) red[tid] = fmaxf(red[tid], red[tid + st]);
    __syncthreads();
  }
  float mx = red[0]; __syncthreads();
  float ex = __expf(sc[tid] - mx);
  sc[tid] = ex; red[tid] = ex; __syncthreads();
  for (int st = PP / 2; st > 0; st >>= 1) {
    if (tid < st) red[tid] += red[tid + st];
    __syncthreads();
  }
  float ssum = red[0]; __syncthreads();

  // context: 256 threads = (e in 0..127) x (p-half); coalesced token reads
  int e = tid & (EE - 1), hp = tid >> 7;
  float acc = 0.f;
  const _Float16* t = tok + (size_t)b * PP * EE + e;
  for (int p = hp * 128; p < hp * 128 + 128; ++p)
    acc += sc[p] * (float)t[(size_t)p * EE];
  red[tid] = acc; __syncthreads();
  if (tid < EE) context[b * EE + tid] = (red[tid] + red[tid + EE]) / ssum;
}

// ---------------------------------------------------------------------------
// K5: qkv = context @ Wqkv + bqkv   (B x 384)
// ---------------------------------------------------------------------------
__global__ void k_qkv(const float* __restrict__ context,
                      const float* __restrict__ Wqkv,
                      const float* __restrict__ bqkv,
                      float* __restrict__ qkv) {
  __shared__ float sctx[EE];
  int b = blockIdx.x, j = threadIdx.x;  // 384 threads
  if (j < EE) sctx[j] = context[b * EE + j];
  __syncthreads();
  float s = bqkv[j];
  for (int k = 0; k < EE; ++k) s += sctx[k] * Wqkv[k * (3 * EE) + j];
  qkv[(size_t)b * (3 * EE) + j] = s;
}

// ---------------------------------------------------------------------------
// K6: tissue histogram (for count = same.sum(-1)); deterministic, no atomics
// ---------------------------------------------------------------------------
__global__ void k_hist(const int* __restrict__ x_tissue, int* __restrict__ hist) {
  int t = threadIdx.x;  // 32 threads
  int c = 0;
  for (int b = 0; b < BB; ++b) c += (x_tissue[b] == t);
  hist[t] = c;
}

// ---------------------------------------------------------------------------
// K7: masked multi-head self-attention over batch: attended[b,:] (B x 128)
// grid B, block 128. Scores for all 2048 columns kept in LDS per head.
// ---------------------------------------------------------------------------
__global__ void k_attn2(const float* __restrict__ qkv,
                        const int*   __restrict__ x_tissue,
                        float* __restrict__ attended) {
  __shared__ float sq[EE];
  __shared__ float ssc[BB];      // 8 KB
  __shared__ float red[EE];
  int b = blockIdx.x, tid = threadIdx.x;
  int tb = x_tissue[b];
  sq[tid] = qkv[(size_t)b * (3 * EE) + tid];  // q row, all heads
  __syncthreads();

  for (int h = 0; h < HH; ++h) {
    for (int c = tid; c < BB; c += EE) {
      float s = -1e9f;
      if (x_tissue[c] == tb) {
        const float* kc = qkv + (size_t)c * (3 * EE) + EE + h * HD;
        float d = 0.f;
        for (int u = 0; u < HD; ++u) d += sq[h * HD + u] * kc[u];
        s = d * 0.17677669529663687f;  // 1/sqrt(32)
      }
      ssc[c] = s;
    }
    __syncthreads();

    float lm = -1e30f;
    for (int c = tid; c < BB; c += EE) lm = fmaxf(lm, ssc[c]);
    red[tid] = lm; __syncthreads();
    for (int st = EE / 2; st > 0; st >>= 1) {
      if (tid < st) red[tid] = fmaxf(red[tid], red[tid + st]);
      __syncthreads();
    }
    float mx = red[0]; __syncthreads();

    float ls = 0.f;
    for (int c = tid; c < BB; c += EE) {
      float e = __expf(ssc[c] - mx);
      ssc[c] = e; ls += e;
    }
    red[tid] = ls; __syncthreads();
    for (int st = EE / 2; st > 0; st >>= 1) {
      if (tid < st) red[tid] += red[tid + st];
      __syncthreads();
    }
    float ssum = red[0]; __syncthreads();

    // attended[b,h,d] = sum_c softmax * v[c,h,d] ; 128 thr = 32 d x 4 quarters
    int d = tid & (HD - 1), q4 = tid >> 5;
    float acc = 0.f;
    for (int c = q4 * (BB / 4); c < (q4 + 1) * (BB / 4); ++c)
      acc += ssc[c] * qkv[(size_t)c * (3 * EE) + 2 * EE + h * HD + d];
    red[tid] = acc; __syncthreads();
    if (tid < HD)
      attended[(size_t)b * EE + h * HD + tid] =
          (red[tid] + red[tid + 32] + red[tid + 64] + red[tid + 96]) / ssum;
    __syncthreads();
  }
}

// ---------------------------------------------------------------------------
// K8: Wo proj + mix + layernorm + count select + MLP head + L2 normalize
// ---------------------------------------------------------------------------
__global__ void k_final(const float* __restrict__ attended,
                        const float* __restrict__ context,
                        const float* __restrict__ Wo, const float* __restrict__ bo,
                        const float* __restrict__ ln_g, const float* __restrict__ ln_b,
                        const float* __restrict__ W1, const float* __restrict__ b1,
                        const float* __restrict__ W2, const float* __restrict__ b2,
                        const int* __restrict__ x_tissue,
                        const int* __restrict__ hist,
                        float* __restrict__ out) {
  __shared__ float sa[EE], sm[EE], red[EE], sh1[32];
  int b = blockIdx.x, tid = threadIdx.x;
  sa[tid] = attended[(size_t)b * EE + tid];
  __syncthreads();
  float s = bo[tid];
  for (int k = 0; k < EE; ++k) s += sa[k] * Wo[k * EE + tid];
  float ctx_in = context[b * EE + tid];
  float pre = 0.3f * s + 0.7f * ctx_in;

  // layernorm over 128
  red[tid] = pre; __syncthreads();
  for (int st = EE / 2; st > 0; st >>= 1) {
    if (tid < st) red[tid] += red[tid + st];
    __syncthreads();
  }
  float mu = red[0] * (1.f / EE); __syncthreads();
  float d0 = pre - mu;
  red[tid] = d0 * d0; __syncthreads();
  for (int st = EE / 2; st > 0; st >>= 1) {
    if (tid < st) red[tid] += red[tid + st];
    __syncthreads();
  }
  float var = red[0] * (1.f / EE); __syncthreads();
  float mixed = d0 * rsqrtf(var + 1e-5f) * ln_g[tid] + ln_b[tid];

  int count = hist[x_tissue[b]];
  sm[tid] = (count > 1) ? mixed : ctx_in;
  __syncthreads();

  if (tid < 32) {
    float h = b1[tid];
    for (int e = 0; e < EE; ++e) h += sm[e] * W1[e * 32 + tid];
    sh1[tid] = fmaxf(h, 0.f);
  }
  __syncthreads();
  if (tid < 2) {
    float l = b2[tid];
    for (int j = 0; j < 32; ++j) l += sh1[j] * W2[j * 2 + tid];
    red[tid] = l;
  }
  __syncthreads();
  if (tid < 2) {
    float n = sqrtf(red[0] * red[0] + red[1] * red[1]);
    out[b * 2 + tid] = red[tid] / fmaxf(n, 1e-12f);
  }
}

// ---------------------------------------------------------------------------
extern "C" void kernel_launch(void* const* d_in, const int* in_sizes, int n_in,
                              void* d_out, int out_size, void* d_ws, size_t ws_size,
                              hipStream_t stream) {
  const float* x_genes     = (const float*)d_in[0];
  const int*   x_tissue    = (const int*)  d_in[1];
  const int*   pathway_idx = (const int*)  d_in[2];
  const float* Wp          = (const float*)d_in[3];
  const float* bp          = (const float*)d_in[4];
  const float* tissue_emb  = (const float*)d_in[5];
  const float* Wq          = (const float*)d_in[6];
  const float* bq          = (const float*)d_in[7];
  const float* Wqkv        = (const float*)d_in[8];
  const float* bqkv        = (const float*)d_in[9];
  const float* Wo          = (const float*)d_in[10];
  const float* bo          = (const float*)d_in[11];
  const float* ln_g        = (const float*)d_in[12];
  const float* ln_b        = (const float*)d_in[13];
  const float* W1          = (const float*)d_in[14];
  const float* b1          = (const float*)d_in[15];
  const float* W2          = (const float*)d_in[16];
  const float* b2          = (const float*)d_in[17];
  float* out = (float*)d_out;

  // workspace layout (bytes)
  char* ws = (char*)d_ws;
  size_t off = 0;
  _Float16* tok   = (_Float16*)(ws + off); off += (size_t)BB * PP * EE * sizeof(_Float16); // 128 MB
  float* tmean    = (float*)(ws + off);    off += (size_t)BB * EE * sizeof(float);
  float* query    = (float*)(ws + off);    off += (size_t)BB * EE * sizeof(float);
  float* context  = (float*)(ws + off);    off += (size_t)BB * EE * sizeof(float);
  float* qkv      = (float*)(ws + off);    off += (size_t)BB * 3 * EE * sizeof(float);
  float* attended = (float*)(ws + off);    off += (size_t)BB * EE * sizeof(float);
  int*   hist     = (int*)  (ws + off);    off += 256;

  k_tokens<<<dim3(BB / 16, PP), 128, 0, stream>>>(x_genes, pathway_idx, Wp, bp, tok);
  k_tmean <<<BB, EE,      0, stream>>>(tok, tmean);
  k_query <<<BB, EE,      0, stream>>>(tmean, tissue_emb, x_tissue, Wq, bq, query);
  k_attnP <<<BB, PP,      0, stream>>>(tok, query, context);
  k_qkv   <<<BB, 3 * EE,  0, stream>>>(context, Wqkv, bqkv, qkv);
  k_hist  <<<1,  TT,      0, stream>>>(x_tissue, hist);
  k_attn2 <<<BB, EE,      0, stream>>>(qkv, x_tissue, attended);
  k_final <<<BB, EE,      0, stream>>>(attended, context, Wo, bo, ln_g, ln_b,
                                       W1, b1, W2, b2, x_tissue, hist, out);
}
